// xAdaptiveAttention_26328149524527
// MI455X (gfx1250) — matrix operand order repaired and checked by reading, not compiled, not run on hardware
//
#include <hip/hip_runtime.h>
#include <hip/hip_bf16.h>
#include <cstdint>

// ---------------------------------------------------------------------------
// MI455X (gfx1250, wave32) implementation of the 5-branch GQA attention block.
// Compute-bound (~137 GFLOP vs ~180MB traffic) -> everything runs on
// v_wmma_f32_16x16x32_bf16 with fp32 accumulation. bf16-resident tiles are
// staged with GLOBAL_LOAD_ASYNC_TO_LDS (ASYNCcnt) when the toolchain has the
// builtins; otherwise we fall back to the vmem->VGPR->ds path.
// ---------------------------------------------------------------------------

typedef unsigned short u16;
typedef __attribute__((ext_vector_type(16))) __bf16 v16bf;
typedef __attribute__((ext_vector_type(8)))  float  v8f;

#if defined(__has_builtin)
#if __has_builtin(__builtin_amdgcn_global_load_async_to_lds_b128)
#define USE_ASYNC_LDS 1
#endif
#endif
#ifndef USE_ASYNC_LDS
#define USE_ASYNC_LDS 0
#endif

#if USE_ASYNC_LDS
typedef int v4i_vs __attribute__((vector_size(16)));
__device__ __forceinline__ void async_ld_b128(const void* g, void* l) {
  __builtin_amdgcn_global_load_async_to_lds_b128(
      (__attribute__((address_space(1))) v4i_vs*)(void*)g,
      (__attribute__((address_space(3))) v4i_vs*)l, 0, 0);
}
__device__ __forceinline__ void async_wait0() {
#if __has_builtin(__builtin_amdgcn_s_wait_asynccnt)
  __builtin_amdgcn_s_wait_asynccnt(0);
#else
  asm volatile("s_wait_asynccnt 0x0" ::: "memory");
#endif
}
#endif

union FragU { v16bf bf; uint4 u[2]; u16 h[16]; };

__device__ __forceinline__ u16 f2bf(float f) {
  union { float f; unsigned u; } c; c.f = f;
  unsigned r = c.u + 0x7FFFu + ((c.u >> 16) & 1u);   // round-to-nearest-even
  return (u16)(r >> 16);
}
__device__ __forceinline__ float bf2f(u16 h) {
  union { unsigned u; float f; } c; c.u = ((unsigned)h) << 16;
  return c.f;
}
__device__ __forceinline__ v8f vzero8() {
  v8f z;
#pragma unroll
  for (int i = 0; i < 8; ++i) z[i] = 0.0f;
  return z;
}
__device__ __forceinline__ v8f wmma_bf16(v16bf a, v16bf b, v8f c) {
  return __builtin_amdgcn_wmma_f32_16x16x32_bf16(false, a, false, b,
                                                 (short)0, c, false, false);
}

// ---------------------------------------------------------------------------
// Generic bf16 WMMA GEMM: C(MxN) = A(MxK) * B(KxN), tile 128x128x32.
// 256 threads = 8 waves (2x4); each wave computes 64x32 via 4x2 16x16 frags.
// AMODE: 0 = A f32 row-major (ptr Af0, ld=Kd)
//        1 = A f32 concat [prev_hidden | hidden[:-1]] (2048+2048 cols)
//        2 = A bf16 row-major (ptr Abf, ld=Kd) -> async LDS copy when available
// EMODE: 0 = f32 store row-major (final output)
//        1 = q scatter: reshape(5,1023,16,128) -> qb[g][h][s][d] bf16
//        2 = kv scatter: k -> kb[kvh][s][d], v -> vt[kvh*128+d][1024] bf16
// ---------------------------------------------------------------------------
template <int AMODE, int EMODE>
__global__ __launch_bounds__(256, 2)
void wmma_gemm(const float* __restrict__ Af0, const float* __restrict__ Af1,
               const u16* __restrict__ Abf,  const float* __restrict__ Bf,
               int M, int N, int Kd,
               u16* __restrict__ qb, u16* __restrict__ kb,
               u16* __restrict__ vtb, float* __restrict__ Co) {
  constexpr int BM = 128, BN = 128, BK = 32, LDT = 40;  // pad 8 halfs
  __shared__ alignas(16) u16 As[2][BM * LDT];           // [m][k]
  __shared__ alignas(16) u16 Bs[2][BN * LDT];           // [n][k] (transposed)

  const int tid  = threadIdx.x;
  const int lane = tid & 31;
  const int wv   = tid >> 5;
  const int wm   = wv >> 2, wn = wv & 3;
  const int m0   = blockIdx.y * BM, n0 = blockIdx.x * BN;

  const int ar = tid >> 1,  ac = (tid & 1) << 4;   // A: 128 rows x 32 cols
  const int bk = tid >> 3,  bn = (tid & 7) << 4;   // B: 32 k-rows x 128 n-cols

  v8f acc[4][2];
#pragma unroll
  for (int i = 0; i < 4; ++i)
#pragma unroll
    for (int j = 0; j < 2; ++j) acc[i][j] = vzero8();

  alignas(16) u16 ta[16];
  alignas(16) u16 tb[16];

  // fetch: issue global loads for k-slice k0; bf16 A goes straight to LDS
  // buffer `nbuf` via async-to-LDS (no VGPR bounce), f32 paths convert in regs.
  auto fetch = [&](int k0, int nbuf) {
    const int rowg = m0 + ar;
    if constexpr (AMODE == 2) {
#if USE_ASYNC_LDS
      const int rr = (rowg < M) ? rowg : (M - 1);  // clamp; row store-guarded
      const u16* ga = Abf + (size_t)rr * Kd + k0 + ac;
      u16* la = &As[nbuf][ar * LDT + ac];
      async_ld_b128(ga, la);
      async_ld_b128(ga + 8, la + 8);
#else
      if (rowg < M) {
        const uint4* p = (const uint4*)(Abf + (size_t)rowg * Kd + k0 + ac);
        ((uint4*)ta)[0] = p[0];
        ((uint4*)ta)[1] = p[1];
      } else {
#pragma unroll
        for (int i = 0; i < 16; ++i) ta[i] = 0;
      }
#endif
    } else {
      if (rowg < M) {
        const float* src;
        const int col = k0 + ac;
        if constexpr (AMODE == 1) {
          src = (col < 2048) ? (Af0 + (size_t)rowg * 2048 + col)
                             : (Af1 + (size_t)rowg * 2048 + (col - 2048));
        } else {
          src = Af0 + (size_t)rowg * Kd + col;
        }
        const float4* p4 = (const float4*)src;
#pragma unroll
        for (int q = 0; q < 4; ++q) {
          float4 f = p4[q];
          ta[q * 4 + 0] = f2bf(f.x); ta[q * 4 + 1] = f2bf(f.y);
          ta[q * 4 + 2] = f2bf(f.z); ta[q * 4 + 3] = f2bf(f.w);
        }
      } else {
#pragma unroll
        for (int i = 0; i < 16; ++i) ta[i] = 0;
      }
    }
    // B is always f32 row-major [Kd][N]: convert + transpose via regs
    {
      const float4* p4 = (const float4*)(Bf + (size_t)(k0 + bk) * N + n0 + bn);
      __builtin_prefetch(Bf + (size_t)(k0 + bk + BK) * N + n0 + bn, 0, 0);
#pragma unroll
      for (int q = 0; q < 4; ++q) {
        float4 f = p4[q];
        tb[q * 4 + 0] = f2bf(f.x); tb[q * 4 + 1] = f2bf(f.y);
        tb[q * 4 + 2] = f2bf(f.z); tb[q * 4 + 3] = f2bf(f.w);
      }
    }
  };

  auto stage = [&](int buf) {
    if constexpr (!(AMODE == 2 && USE_ASYNC_LDS)) {
      uint4* d = (uint4*)&As[buf][ar * LDT + ac];
      d[0] = ((uint4*)ta)[0];
      d[1] = ((uint4*)ta)[1];
    }
#pragma unroll
    for (int i = 0; i < 16; ++i) Bs[buf][(bn + i) * LDT + bk] = tb[i];  // transpose
  };

  auto compute = [&](int buf) {
    const u16* Ab = As[buf];
    const u16* Bb = Bs[buf];
    const int abase = (lane >= 16) ? 8 : 0;   // A frag: halves 0-7=K b..b+7, 8-15=K 16+b..
    const int bbase = (lane >= 16) ? 8 : 0;   // B frag: halves 0-7=K b..b+7, 8-15=K 16+b..
    v16bf af[4], bfv[2];
#pragma unroll
    for (int mi = 0; mi < 4; ++mi) {
      FragU u;
      const u16* p = Ab + (wm * 64 + mi * 16 + (lane & 15)) * LDT;
      u.u[0] = *(const uint4*)(p + abase);
      u.u[1] = *(const uint4*)(p + abase + 16);
      af[mi] = u.bf;
    }
#pragma unroll
    for (int ni = 0; ni < 2; ++ni) {
      FragU u;
      const u16* p = Bb + (wn * 32 + ni * 16 + (lane & 15)) * LDT;
      u.u[0] = *(const uint4*)(p + bbase);
      u.u[1] = *(const uint4*)(p + bbase + 16);
      bfv[ni] = u.bf;
    }
#pragma unroll
    for (int mi = 0; mi < 4; ++mi)
#pragma unroll
      for (int ni = 0; ni < 2; ++ni)
        acc[mi][ni] = wmma_bf16(af[mi], bfv[ni], acc[mi][ni]);
  };

  const int nk = Kd / BK;
  fetch(0, 0);
  stage(0);
#if USE_ASYNC_LDS
  if constexpr (AMODE == 2) async_wait0();
#endif
  __syncthreads();
  for (int kt = 0; kt < nk; ++kt) {
    if (kt + 1 < nk) fetch((kt + 1) * BK, (kt + 1) & 1);
    compute(kt & 1);
    __syncthreads();
    if (kt + 1 < nk) {
      stage((kt + 1) & 1);
#if USE_ASYNC_LDS
      if constexpr (AMODE == 2) async_wait0();
#endif
    }
    __syncthreads();
  }

  // Epilogue. C layout: lane<16 -> M=v, lane>=16 -> M=8+v ; N = lane&15.
#pragma unroll
  for (int mi = 0; mi < 4; ++mi) {
    const int rb = m0 + wm * 64 + mi * 16 + ((lane >> 4) << 3);
#pragma unroll
    for (int ni = 0; ni < 2; ++ni) {
      const int col = n0 + wn * 32 + ni * 16 + (lane & 15);
#pragma unroll
      for (int v = 0; v < 8; ++v) {
        const int row = rb + v;
        if (row < M) {
          const float val = acc[mi][ni][v];
          if constexpr (EMODE == 0) {
            Co[(size_t)row * N + col] = val;
          } else if constexpr (EMODE == 1) {
            // q flat reindex: (5,1023,16,128) from flat s*10240 + n
            const unsigned f  = (unsigned)row * 10240u + (unsigned)col;
            const int g  = (int)(f / 2095104u);           // 1023*2048
            const int r  = (int)(f % 2095104u);
            const int s2 = r >> 11;
            const int c2 = r & 2047;
            qb[(((size_t)g * 16 + (c2 >> 7)) * 1023 + s2) * 128 + (c2 & 127)] =
                f2bf(val);
          } else {
            if (col < 512)
              kb[((size_t)(col >> 7) * 1023 + row) * 128 + (col & 127)] = f2bf(val);
            else
              vtb[(size_t)(col - 512) * 1024 + row] = f2bf(val);  // [kvh*128+d][s]
          }
        }
      }
    }
  }
}

// ---------------------------------------------------------------------------
// RoPE over rows of [nh][1023][128] bf16: pair (j, j+64) per thread, in-place.
// ---------------------------------------------------------------------------
__global__ void rope_kernel(u16* __restrict__ buf, int nh) {
  const int idx = blockIdx.x * blockDim.x + threadIdx.x;
  if (idx >= nh * 1023 * 64) return;
  const int j  = idx & 63;
  const int s  = (idx >> 6) % 1023;
  const int hh = idx / (64 * 1023);
  const size_t base = ((size_t)hh * 1023 + s) * 128;
  const float x0 = bf2f(buf[base + j]);
  const float x1 = bf2f(buf[base + j + 64]);
  const float inv = __expf(-0.14391156516f * (float)j);  // ln(1e4)/64
  float sn, cs;
  __sincosf((float)s * inv, &sn, &cs);
  buf[base + j]      = f2bf(x0 * cs - x1 * sn);
  buf[base + j + 64] = f2bf(x1 * cs + x0 * sn);
}

// ---------------------------------------------------------------------------
// Flash attention. Grid: x = g*16+h (80), y = q-block of 64 rows (16).
// 128 threads = 4 waves; wave w owns 16 q rows; K/V 32-wide tiles in LDS
// (async global->LDS when available; rows >= kv_len are masked to exp->0).
// ---------------------------------------------------------------------------
__global__ __launch_bounds__(128, 2)
void attn_kernel(const u16* __restrict__ qb, const u16* __restrict__ kb,
                 const u16* __restrict__ vtb, u16* __restrict__ ab) {
  constexpr int KL = 136, VL = 40;
  __shared__ alignas(16) u16 Ks[32 * KL];    // [kv][d]
  __shared__ alignas(16) u16 Vs[128 * VL];   // [d][kv]
  __shared__ alignas(16) u16 Ps[4][16 * 32]; // per-wave P staging

  const int gh = blockIdx.x;
  const int g = gh >> 4, h = gh & 15, kvh = h >> 2;
  const int tid = threadIdx.x, lane = tid & 31, w = tid >> 5;
  const int qbase = blockIdx.y * 64;
  const int qrow0 = qbase + w * 16;

  // Q fragments: A-matrix 16x128 -> 4 chunks of K=32
  v16bf qf[4];
  {
    int srow = qrow0 + (lane & 15);
    if (srow > 1022) srow = 1022;  // clamped row is masked out at store
    const u16* qr = qb + ((size_t)gh * 1023 + srow) * 128;
    const int abase = (lane >= 16) ? 8 : 0;
#pragma unroll
    for (int c = 0; c < 4; ++c) {
      FragU u;
      u.u[0] = *(const uint4*)(qr + c * 32 + abase);
      u.u[1] = *(const uint4*)(qr + c * 32 + abase + 16);
      qf[c] = u.bf;
    }
  }

  float mst[8], lst[8];
  v8f oc[8];
#pragma unroll
  for (int v = 0; v < 8; ++v) { mst[v] = -1e30f; lst[v] = 0.0f; }
#pragma unroll
  for (int n = 0; n < 8; ++n) oc[n] = vzero8();

  const int jtmax = (qbase + 63) >> 5;
  for (int jt = 0; jt <= jtmax; ++jt) {
    const int jb = jt * 32;
    {  // K tile [32][128]
      const int r = tid >> 2, cb = (tid & 3) << 5;
#if USE_ASYNC_LDS
      // clamp-free async copy; kv row 1023 is always softmax-masked
      const u16* gk = kb + ((size_t)kvh * 1023 + jb + r) * 128 + cb;
      u16* lk = &Ks[r * KL + cb];
      async_ld_b128(gk, lk);
      async_ld_b128(gk + 8,  lk + 8);
      async_ld_b128(gk + 16, lk + 16);
      async_ld_b128(gk + 24, lk + 24);
#else
      uint4 d0, d1, d2, d3;
      if (jb + r < 1023) {
        const uint4* p = (const uint4*)(kb + ((size_t)kvh * 1023 + jb + r) * 128 + cb);
        d0 = p[0]; d1 = p[1]; d2 = p[2]; d3 = p[3];
      } else {
        d0.x = d0.y = d0.z = d0.w = 0u; d1 = d0; d2 = d0; d3 = d0;
      }
      uint4* q4 = (uint4*)&Ks[r * KL + cb];
      q4[0] = d0; q4[1] = d1; q4[2] = d2; q4[3] = d3;
#endif
    }
    {  // V^T tile [128][32] (vt padded to ld=1024; masked cols contribute 0)
      const u16* gv = vtb + ((size_t)kvh * 128 + tid) * 1024 + jb;
#if USE_ASYNC_LDS
      u16* lv = &Vs[tid * VL];
      async_ld_b128(gv, lv);
      async_ld_b128(gv + 8,  lv + 8);
      async_ld_b128(gv + 16, lv + 16);
      async_ld_b128(gv + 24, lv + 24);
#else
      const uint4* p = (const uint4*)gv;
      uint4* q4 = (uint4*)&Vs[tid * VL];
      q4[0] = p[0]; q4[1] = p[1]; q4[2] = p[2]; q4[3] = p[3];
#endif
    }
#if USE_ASYNC_LDS
    async_wait0();
#endif
    __syncthreads();

    if (jb <= qrow0 + 15) {  // wave-uniform causal skip (EXEC all-1s for WMMA)
      v8f s0 = vzero8(), s1 = vzero8();
      const int kb0 = (lane >= 16) ? 8 : 0;
#pragma unroll
      for (int c = 0; c < 4; ++c) {  // S = Q K^T
        FragU u0, u1;
        const u16* p0 = &Ks[(lane & 15) * KL + c * 32 + kb0];
        u0.u[0] = *(const uint4*)p0; u0.u[1] = *(const uint4*)(p0 + 16);
        const u16* p1 = &Ks[((lane & 15) + 16) * KL + c * 32 + kb0];
        u1.u[0] = *(const uint4*)p1; u1.u[1] = *(const uint4*)(p1 + 16);
        s0 = wmma_bf16(qf[c], u0.bf, s0);
        s1 = wmma_bf16(qf[c], u1.bf, s1);
      }
      // online softmax (rows live per-VGPR; cols across 16 lanes of a half)
      const float sc = 0.088388347648318447f;  // 1/sqrt(128)
      const int c0 = jb + (lane & 15), c1 = c0 + 16;
      const int rbase = qrow0 + ((lane >> 4) << 3);
      float p0a[8], p1a[8];
#pragma unroll
      for (int v = 0; v < 8; ++v) {
        const int rr = rbase + v;
        float a0 = s0[v] * sc + ((c0 <= rr && c0 < 1023) ? 0.0f : -1e9f);
        float a1 = s1[v] * sc + ((c1 <= rr && c1 < 1023) ? 0.0f : -1e9f);
        float mv = fmaxf(a0, a1);
#pragma unroll
        for (int off = 8; off >= 1; off >>= 1) mv = fmaxf(mv, __shfl_xor(mv, off));
        const float mn = fmaxf(mst[v], mv);
        const float al = __expf(mst[v] - mn);
        const float e0 = __expf(a0 - mn);
        const float e1 = __expf(a1 - mn);
        float rs = e0 + e1;
#pragma unroll
        for (int off = 8; off >= 1; off >>= 1) rs += __shfl_xor(rs, off);
        lst[v] = lst[v] * al + rs;
        mst[v] = mn;
#pragma unroll
        for (int n = 0; n < 8; ++n) oc[n][v] *= al;
        p0a[v] = e0; p1a[v] = e1;
      }
      // stage P (C layout) -> LDS -> reload in A layout (per-wave, LDS in-order)
#pragma unroll
      for (int v = 0; v < 8; ++v) {
        const int pr = ((lane >> 4) << 3) + v;
        Ps[w][pr * 32 + (lane & 15)]      = f2bf(p0a[v]);
        Ps[w][pr * 32 + 16 + (lane & 15)] = f2bf(p1a[v]);
      }
      __builtin_amdgcn_wave_barrier();
      FragU pu;
      const int pb = (lane >= 16) ? 8 : 0;
      const u16* pp = &Ps[w][(lane & 15) * 32 + pb];
      pu.u[0] = *(const uint4*)pp;
      pu.u[1] = *(const uint4*)(pp + 16);
      const int vb = (lane >= 16) ? 8 : 0;
#pragma unroll
      for (int n = 0; n < 8; ++n) {  // O += P V
        FragU vu;
        const u16* vp = &Vs[(n * 16 + (lane & 15)) * VL + vb];
        vu.u[0] = *(const uint4*)vp; vu.u[1] = *(const uint4*)(vp + 16);
        oc[n] = wmma_bf16(pu.bf, vu.bf, oc[n]);
      }
    }
    __syncthreads();
  }

  // write attn output as bf16 rows of [g*1023+s][h*128+d] for the final GEMM
#pragma unroll
  for (int n = 0; n < 8; ++n)
#pragma unroll
    for (int v = 0; v < 8; ++v) {
      const int row = qrow0 + ((lane >> 4) << 3) + v;
      if (row < 1023)
        ab[((size_t)g * 1023 + row) * 2048 + h * 128 + n * 16 + (lane & 15)] =
            f2bf(oc[n][v] / lst[v]);
    }
}

// ---------------------------------------------------------------------------
extern "C" void kernel_launch(void* const* d_in, const int* in_sizes, int n_in,
                              void* d_out, int out_size, void* d_ws, size_t ws_size,
                              hipStream_t stream) {
  (void)in_sizes; (void)n_in; (void)out_size; (void)ws_size;
  const float* hidden = (const float*)d_in[0];  // (1,1024,2048)
  const float* prevh  = (const float*)d_in[1];  // (1,1023,2048)
  const float* Wq     = (const float*)d_in[4];  // (2048,10240)
  const float* Wkv    = (const float*)d_in[5];  // (4096,1024)
  const float* Wo     = (const float*)d_in[6];  // (2048,2048)
  float* out = (float*)d_out;                   // (5,1023,2048)

  char* ws = (char*)d_ws;
  u16* qb   = (u16*)ws; ws += (size_t)80 * 1023 * 128 * 2;  // [g][h][s][d]
  u16* kbuf = (u16*)ws; ws += (size_t)4 * 1023 * 128 * 2;   // [kvh][s][d]
  u16* vtb  = (u16*)ws; ws += (size_t)4 * 128 * 1024 * 2;   // [kvh*128+d][1024]
  u16* ab   = (u16*)ws;                                     // [5115][2048]

  // q = hs @ Wq  (hs = hidden[:,1:]), fused reshape-scatter to qb
  wmma_gemm<0, 1><<<dim3(80, 8), 256, 0, stream>>>(
      hidden + 2048, nullptr, nullptr, Wq, 1023, 10240, 2048,
      qb, nullptr, nullptr, nullptr);
  // kv = [prev_hidden | hidden[:-1]] @ Wkv, fused k/v split (v transposed)
  wmma_gemm<1, 2><<<dim3(8, 8), 256, 0, stream>>>(
      prevh, hidden, nullptr, Wkv, 1023, 1024, 4096,
      nullptr, kbuf, vtb, nullptr);
  // RoPE on q (80 head-rows) and k (4 kv-head rows)
  rope_kernel<<<(80 * 1023 * 64 + 255) / 256, 256, 0, stream>>>(qb, 80);
  rope_kernel<<<(4 * 1023 * 64 + 255) / 256, 256, 0, stream>>>(kbuf, 4);
  // flash attention over 80 head-problems
  attn_kernel<<<dim3(80, 16), 128, 0, stream>>>(qb, kbuf, vtb, ab);
  // out = attn_out @ Wo
  wmma_gemm<2, 0><<<dim3(16, 40), 256, 0, stream>>>(
      nullptr, nullptr, ab, Wo, 5115, 2048, 2048,
      nullptr, nullptr, nullptr, out);
}
